// OurAdapterLayer_52029233824452
// MI455X (gfx1250) — matrix operand order repaired and checked by reading, not compiled, not run on hardware
//
#include <hip/hip_runtime.h>

#define B_ 32
#define T_ 1024
#define D_ 1024
#define R_ 64
#define E_ 8

typedef __bf16 v16bf  __attribute__((ext_vector_type(16)));
typedef __bf16 bf16x8 __attribute__((ext_vector_type(8)));
typedef float  v8f    __attribute__((ext_vector_type(8)));

__device__ __forceinline__ v16bf cat8(bf16x8 lo, bf16x8 hi) {
    return __builtin_shufflevector(lo, hi, 0,1,2,3,4,5,6,7,8,9,10,11,12,13,14,15);
}
__device__ __forceinline__ bf16x8 cvt8(const float* p) {
    float4 a = *(const float4*)p;
    float4 b = *(const float4*)(p + 4);
    bf16x8 r;
    r[0] = (__bf16)a.x; r[1] = (__bf16)a.y; r[2] = (__bf16)a.z; r[3] = (__bf16)a.w;
    r[4] = (__bf16)b.x; r[5] = (__bf16)b.y; r[6] = (__bf16)b.z; r[7] = (__bf16)b.w;
    return r;
}

// ---- K0: bulk f32 -> bf16 conversion (8 elements / thread) ---------------
__global__ __launch_bounds__(256) void cvt_bf16_kernel(const float* __restrict__ src,
                                                       __bf16* __restrict__ dst) {
    size_t i = ((size_t)blockIdx.x * 256 + threadIdx.x) * 8;
    *(bf16x8*)(dst + i) = cvt8(src + i);
}

// ---- K1: per-(b,d) sums over T ------------------------------------------
__global__ __launch_bounds__(256) void stats_kernel(const float* __restrict__ x,
                                                    float* __restrict__ m1s,
                                                    float* __restrict__ m2s) {
    int b = blockIdx.x >> 2;
    int d = ((blockIdx.x & 3) << 8) + threadIdx.x;
    const float* p = x + (size_t)b * T_ * D_ + d;
    float s1 = 0.f, s2 = 0.f;
    for (int t = 0; t < T_; ++t) { float v = p[(size_t)t * D_]; s1 += v; s2 += v * v; }
    m1s[b * D_ + d] = s1;
    m2s[b * D_ + d] = s2;
}

// ---- K2: losses [E,B] + argmin -> idx[B] --------------------------------
__global__ __launch_bounds__(256) void route_kernel(const float* __restrict__ m1s,
                                                    const float* __restrict__ m2s,
                                                    const float* __restrict__ mu,
                                                    const float* __restrict__ sigma,
                                                    float* __restrict__ losses,
                                                    int* __restrict__ idx) {
    __shared__ float ls[E_][B_];
    int e = threadIdx.x >> 5;
    int b = threadIdx.x & 31;
    const float invT = 1.0f / (float)T_;
    float acc = 0.f;
    for (int d = 0; d < D_; ++d) {
        float sg = sigma[e * D_ + d];
        float iv = 1.0f / (sg * sg);
        float m  = mu[e * D_ + d];
        float m1 = m1s[b * D_ + d] * invT;
        float m2 = m2s[b * D_ + d] * invT;
        acc += (m2 - 2.f * m1 * m + m * m) * iv;
    }
    float loss = acc / (float)D_;
    ls[e][b] = loss;
    losses[e * B_ + b] = loss;
    __syncthreads();
    if (threadIdx.x < B_) {
        int bb = threadIdx.x;
        float best = ls[0][bb]; int bi = 0;
        for (int ee = 1; ee < E_; ++ee) {
            float v = ls[ee][bb];
            if (v < best) { best = v; bi = ee; }
        }
        idx[bb] = bi;
    }
}

// ---- K3: hbf[b,t,r] = bf16(relu(x @ down_W[idx] + down_b[idx])) ----------
__global__ __launch_bounds__(256) void down_kernel(const float* __restrict__ x,
                                                   const float* __restrict__ down_W,
                                                   const float* __restrict__ down_b,
                                                   const int* __restrict__ idx,
                                                   __bf16* __restrict__ hbf) {
    __shared__ float xs[4][D_];
    int b  = blockIdx.x >> 8;
    int t0 = (blockIdx.x & 255) << 2;
    int e  = idx[b];
    const float* xp = x + ((size_t)b * T_ + t0) * D_;
    for (int i = threadIdx.x; i < 1024; i += 256) {
        int row = i >> 8;
        int c   = (i & 255) << 2;
        *(float4*)&xs[row][c] = *(const float4*)(xp + (size_t)row * D_ + c);
    }
    __syncthreads();
    int tl = threadIdx.x >> 6;
    int r  = threadIdx.x & 63;
    const float* w = down_W + (size_t)e * D_ * R_ + r;
    float acc = down_b[e * R_ + r];
    for (int d = 0; d < D_; ++d) acc += xs[tl][d] * w[(size_t)d * R_];
    hbf[((size_t)b * T_ + t0 + tl) * R_ + r] = (__bf16)fmaxf(acc, 0.f);
}

// ---- K4a: Mtbf[b][o][r] = bf16(sum_d Wc[e][o][d] * Wu[e][r][d]) (WMMA) ---
__global__ __launch_bounds__(256) void fuseW_kernel(const float* __restrict__ up_W,
                                                    const float* __restrict__ conv_W,
                                                    const int* __restrict__ idx,
                                                    __bf16* __restrict__ Mtbf) {
    int wid  = threadIdx.x >> 5;
    int lane = threadIdx.x & 31;
    int gid  = blockIdx.x * 8 + wid;
    int b    = gid >> 6;
    int o0   = (gid & 63) << 4;
    int e    = idx[b];
    int half = lane >> 4;
    int l16  = lane & 15;
    const float* Wc = conv_W + (size_t)e * D_ * D_;
    const float* Wu = up_W   + (size_t)e * R_ * D_;
    const float* arow = Wc + (size_t)(o0 + l16) * D_;
    v8f zero = {0.f,0.f,0.f,0.f,0.f,0.f,0.f,0.f};
    v8f acc[4] = {zero, zero, zero, zero};
    for (int d0 = 0; d0 < D_; d0 += 32) {
        v16bf a = cat8(cvt8(arow + d0 + half * 16), cvt8(arow + d0 + half * 16 + 8));
#pragma unroll
        for (int fn = 0; fn < 4; ++fn) {
            const float* bp = Wu + (size_t)(fn * 16 + l16) * D_ + d0 + half * 16;
            v16bf bb = cat8(cvt8(bp), cvt8(bp + 8));
            acc[fn] = __builtin_amdgcn_wmma_f32_16x16x32_bf16(
                false, a, false, bb, (short)0, acc[fn], false, false);
        }
    }
#pragma unroll
    for (int fn = 0; fn < 4; ++fn)
#pragma unroll
        for (int v = 0; v < 8; ++v) {
            int o = o0 + v + half * 8;
            int r = fn * 16 + l16;
            Mtbf[((size_t)b * D_ + o) * R_ + r] = (__bf16)acc[fn][v];
        }
}

// ---- K4b: cvec[b][o] = base_b[o] + conv_b[e][o] + sum_d up_b[e][d]*Wc[e][o][d]
__global__ __launch_bounds__(256) void cvec_kernel(const float* __restrict__ base_b,
                                                   const float* __restrict__ up_b,
                                                   const float* __restrict__ conv_W,
                                                   const float* __restrict__ conv_b,
                                                   const int* __restrict__ idx,
                                                   float* __restrict__ cvec) {
    int b = blockIdx.x >> 2;
    int o = ((blockIdx.x & 3) << 8) + threadIdx.x;
    int e = idx[b];
    const float* wr = conv_W + ((size_t)e * D_ + o) * D_;
    const float* bu = up_b + e * D_;
    float acc = base_b[o] + conv_b[e * D_ + o];
    for (int d = 0; d < D_; ++d) acc += bu[d] * wr[d];
    cvec[b * D_ + o] = acc;
}

// ---- K5: out = x@Wb^T + h@Mt^T + cvec  (bf16 WMMA, async-to-LDS staging) -
// 128x128 tile / block, BK=32, 8 waves (4m x 2n), wave = 32x64 (2x4 frags).
// Double-buffered LDS; next stage loaded with global_load_async_to_lds_b128.
__global__ __launch_bounds__(256) void main_gemm(const __bf16* __restrict__ xbf,
                                                 const __bf16* __restrict__ Wbf,
                                                 const __bf16* __restrict__ hbf,
                                                 const __bf16* __restrict__ Mtbf,
                                                 const float* __restrict__ cvec,
                                                 float* __restrict__ out) {
    __shared__ __bf16 tiles[2][2][128 * 32];   // [buf][A=0 / B=1]

    int bid = blockIdx.x;
    int b   = bid >> 6;
    int t0  = ((bid >> 3) & 7) << 7;
    int n0  = (bid & 7) << 7;
    int tid  = threadIdx.x;
    int wid  = tid >> 5, lane = tid & 31;
    int half = lane >> 4, l16 = lane & 15;
    int wm = (wid & 3) * 32;
    int wn = (wid >> 2) * 64;

    v8f zero = {0.f,0.f,0.f,0.f,0.f,0.f,0.f,0.f};
    v8f acc[2][4];
#pragma unroll
    for (int i = 0; i < 2; ++i)
#pragma unroll
        for (int j = 0; j < 4; ++j) acc[i][j] = zero;

    const __bf16* xA = xbf  + ((size_t)b * T_ + t0) * D_;
    const __bf16* hA = hbf  + ((size_t)b * T_ + t0) * R_;
    const __bf16* wB = Wbf  + (size_t)n0 * D_;
    const __bf16* mB = Mtbf + ((size_t)b * D_ + n0) * R_;

    // each thread moves 4 x 16B chunks per stage: rows (row0, row0+64), col c8
    int row0 = tid >> 2;
    int c8   = (tid & 3) * 8;

    auto params = [&](int s, const __bf16*& sA, int& strA,
                      const __bf16*& sB, int& strB, int& k0) {
        if (s < 32) { sA = xA; strA = D_; sB = wB; strB = D_; k0 = s * 32; }
        else        { sA = hA; strA = R_; sB = mB; strB = R_; k0 = (s - 32) * 32; }
    };
    auto issue = [&](int s, int buf) {
        const __bf16 *sA, *sB; int strA, strB, k0;
        params(s, sA, strA, sB, strB, k0);
        const __bf16* gA0 = sA + (size_t)row0 * strA + k0 + c8;
        const __bf16* gA1 = sA + (size_t)(row0 + 64) * strA + k0 + c8;
        const __bf16* gB0 = sB + (size_t)row0 * strB + k0 + c8;
        const __bf16* gB1 = sB + (size_t)(row0 + 64) * strB + k0 + c8;
        unsigned lA0 = (unsigned)(size_t)&tiles[buf][0][row0 * 32 + c8];
        unsigned lA1 = (unsigned)(size_t)&tiles[buf][0][(row0 + 64) * 32 + c8];
        unsigned lB0 = (unsigned)(size_t)&tiles[buf][1][row0 * 32 + c8];
        unsigned lB1 = (unsigned)(size_t)&tiles[buf][1][(row0 + 64) * 32 + c8];
        asm volatile("global_load_async_to_lds_b128 %0, %1, off" :: "v"(lA0), "v"(gA0) : "memory");
        asm volatile("global_load_async_to_lds_b128 %0, %1, off" :: "v"(lA1), "v"(gA1) : "memory");
        asm volatile("global_load_async_to_lds_b128 %0, %1, off" :: "v"(lB0), "v"(gB0) : "memory");
        asm volatile("global_load_async_to_lds_b128 %0, %1, off" :: "v"(lB1), "v"(gB1) : "memory");
    };

    issue(0, 0);   // prologue

    for (int s = 0; s < 34; ++s) {
        int buf = s & 1;
        if (s + 1 < 34) {
            issue(s + 1, buf ^ 1);
            // 4 newer requests outstanding; in-order completion => stage s done
            asm volatile("s_wait_asynccnt 4" ::: "memory");
        } else {
            asm volatile("s_wait_asynccnt 0" ::: "memory");
        }
        __syncthreads();

        const __bf16* As = &tiles[buf][0][0];
        const __bf16* Bs = &tiles[buf][1][0];
#pragma unroll
        for (int fm = 0; fm < 2; ++fm) {
            const __bf16* ap = As + (wm + fm * 16 + l16) * 32 + half * 16;
            v16bf a = cat8(*(const bf16x8*)ap, *(const bf16x8*)(ap + 8));
#pragma unroll
            for (int fn = 0; fn < 4; ++fn) {
                const __bf16* bp = Bs + (wn + fn * 16 + l16) * 32 + half * 16;
                v16bf bb = *(const v16bf*)bp;
                acc[fm][fn] = __builtin_amdgcn_wmma_f32_16x16x32_bf16(
                    false, a, false, bb, (short)0, acc[fm][fn], false, false);
            }
        }
        __syncthreads();
    }

    float cv[4];
#pragma unroll
    for (int fn = 0; fn < 4; ++fn) cv[fn] = cvec[b * D_ + n0 + wn + fn * 16 + l16];
#pragma unroll
    for (int fm = 0; fm < 2; ++fm)
#pragma unroll
        for (int fn = 0; fn < 4; ++fn)
#pragma unroll
            for (int v = 0; v < 8; ++v) {
                int t = t0 + wm + fm * 16 + v + half * 8;
                int o = n0 + wn + fn * 16 + l16;
                out[((size_t)b * T_ + t) * D_ + o] = acc[fm][fn][v] + cv[fn];
            }
}

// ---- launcher ------------------------------------------------------------
extern "C" void kernel_launch(void* const* d_in, const int* in_sizes, int n_in,
                              void* d_out, int out_size, void* d_ws, size_t ws_size,
                              hipStream_t stream) {
    const float* x          = (const float*)d_in[0];
    const float* base_W     = (const float*)d_in[1];
    const float* base_b     = (const float*)d_in[2];
    const float* disc_mu    = (const float*)d_in[3];
    const float* disc_sigma = (const float*)d_in[4];
    const float* down_W     = (const float*)d_in[5];
    const float* down_b     = (const float*)d_in[6];
    const float* up_W       = (const float*)d_in[7];
    const float* up_b       = (const float*)d_in[8];
    const float* conv_W     = (const float*)d_in[9];
    const float* conv_b     = (const float*)d_in[10];
    float* out = (float*)d_out;

    // workspace layout
    char*  wsb    = (char*)d_ws;
    float* m1s    = (float*)wsb;            // 32768 f32
    float* m2s    = m1s + 32768;            // 32768 f32
    float* losses = m1s + 65536;            // 256 f32
    int*   idx    = (int*)(m1s + 65792);    // 32 i32
    float* cvec   = m1s + 66048;            // 32768 f32  (f32 region = 395264 B)
    __bf16* xbf   = (__bf16*)(wsb + 395264);            // B*T*D  bf16 (64 MB)
    __bf16* Wbf   = xbf + (size_t)B_ * T_ * D_;         // D*D    bf16 (2 MB)
    __bf16* hbf   = Wbf + (size_t)D_ * D_;              // B*T*R  bf16 (4 MB)
    __bf16* Mtbf  = hbf + (size_t)B_ * T_ * R_;         // B*D*R  bf16 (4 MB)

    cvt_bf16_kernel<<<(B_ * T_ * D_) / 2048, 256, 0, stream>>>(x, xbf);
    cvt_bf16_kernel<<<(D_ * D_) / 2048, 256, 0, stream>>>(base_W, Wbf);
    stats_kernel<<<B_ * 4, 256, 0, stream>>>(x, m1s, m2s);
    route_kernel<<<1, 256, 0, stream>>>(m1s, m2s, disc_mu, disc_sigma, losses, idx);
    down_kernel<<<B_ * (T_ / 4), 256, 0, stream>>>(x, down_W, down_b, idx, hbf);
    fuseW_kernel<<<B_ * 64 / 8, 256, 0, stream>>>(up_W, conv_W, idx, Mtbf);
    cvec_kernel<<<B_ * 4, 256, 0, stream>>>(base_b, up_b, conv_W, conv_b, idx, cvec);
    main_gemm<<<B_ * 64, 256, 0, stream>>>(xbf, Wbf, hbf, Mtbf, cvec, out);
}
